// DilatedAttention_83829171683597
// MI455X (gfx1250) — compile-verified, matrix operand order checked by reading, not run on hardware
//
#include <hip/hip_runtime.h>
#include <hip/hip_bf16.h>

// CDNA5 / gfx1250 dilated attention (LongNet style), bf16 WMMA, f32 accum.
// x: [4, 8192, 1024] f32.  r=4, seg=512 -> 64 independent 512x512 attention
// problems over d_model=1024; slice rows strided by r*D floats in memory.
//
// Fast path (needs 128 MiB workspace, L2-resident on MI455X's 192 MB L2):
//   prepass: x(f32) -> xb bf16 row-major  [B,S,D]
//                   -> xt bf16 transposed [B*r, D, S/r]   (LDS tile transpose)
//   main:    phase 1  S = Q K^T   from xb   (pure b128 loads, no cvt)
//            phase 2  softmax from registers (ds_swizzle xor reductions)
//            phase 3  O^T = V^T P^T from xt + LDS P (1 b128 load per WMMA)
// Fallback path (no workspace): self-contained kernel with inline f32->bf16.

typedef __attribute__((ext_vector_type(16))) __bf16 v16bf;
typedef __attribute__((ext_vector_type(8)))  __bf16 v8bf;
typedef __attribute__((ext_vector_type(8)))  float  v8f;

#define B_    4
#define S_    8192
#define D_    1024
#define R_    4
#define SEG_  512
#define NSL_  2048                       // S_/R_: rows per strided slice
#define ROWSTRIDE ((size_t)(R_ * D_))    // elements between consecutive slice rows
#define SCALE (0.03125f)                 // 1/sqrt(1024)
#define PM_LD 520                        // bf16 prob row stride (512+8); 1040B = 65*16

#define WMMA_BF16(A, B, C) __builtin_amdgcn_wmma_f32_16x16x32_bf16( \
        false, (A), false, (B), (short)0, (C), false, false)

// ---- lane-group (16) xor reductions via ds_swizzle (and=0x1F, xor=mk<<10) ----
static __device__ __forceinline__ float swz_max16(float v) {
    v = fmaxf(v, __int_as_float(__builtin_amdgcn_ds_swizzle(__float_as_int(v), 0x1F | (1 << 10))));
    v = fmaxf(v, __int_as_float(__builtin_amdgcn_ds_swizzle(__float_as_int(v), 0x1F | (2 << 10))));
    v = fmaxf(v, __int_as_float(__builtin_amdgcn_ds_swizzle(__float_as_int(v), 0x1F | (4 << 10))));
    v = fmaxf(v, __int_as_float(__builtin_amdgcn_ds_swizzle(__float_as_int(v), 0x1F | (8 << 10))));
    return v;
}
static __device__ __forceinline__ float swz_sum16(float v) {
    v += __int_as_float(__builtin_amdgcn_ds_swizzle(__float_as_int(v), 0x1F | (1 << 10)));
    v += __int_as_float(__builtin_amdgcn_ds_swizzle(__float_as_int(v), 0x1F | (2 << 10)));
    v += __int_as_float(__builtin_amdgcn_ds_swizzle(__float_as_int(v), 0x1F | (4 << 10)));
    v += __int_as_float(__builtin_amdgcn_ds_swizzle(__float_as_int(v), 0x1F | (8 << 10)));
    return v;
}

// Build a 16x32 bf16 fragment from two contiguous 8-element chunks at p, p+16.
static __device__ __forceinline__ v16bf ldfrag(const __bf16* p) {
    v8bf lo = *(const v8bf*)p;
    v8bf hi = *(const v8bf*)(p + 16);
    return __builtin_shufflevector(lo, hi, 0, 1, 2, 3, 4, 5, 6, 7,
                                           8, 9, 10, 11, 12, 13, 14, 15);
}

// ======================= pre-pass: f32 -> bf16 (+transpose) ==================
// grid: 16 (b,off) x 32 i-tiles x 16 d-tiles = 8192 blocks, 256 threads.
// Each block handles a 64(i) x 64(d) tile of one strided slice.
__global__ __launch_bounds__(256)
void prepass_kernel(const float* __restrict__ x,
                    __bf16* __restrict__ xb, __bf16* __restrict__ xt) {
    __shared__ __bf16 t[64][72];                 // 72*2=144B row stride (16B mult.)

    const int blk  = blockIdx.x;
    const int bo   = blk >> 9;                   // 0..15 = b*4+off
    const int b    = bo >> 2, off = bo & 3;
    const int trem = blk & 511;
    const int it   = trem >> 4, dt = trem & 15;
    const int ibase = it * 64, dbase = dt * 64;

    const int tid = threadIdx.x;
    const int row = tid >> 2;                    // 0..63 local slice-row
    const int cs  = (tid & 3) * 16;              // 16-col segment

    const size_t s = (size_t)(ibase + row) * R_ + off;
    const float* src = x + ((size_t)b * S_ + s) * D_ + dbase + cs;

    v8bf v0, v1;
    #pragma unroll
    for (int j = 0; j < 8; ++j) { v0[j] = (__bf16)src[j]; v1[j] = (__bf16)src[8 + j]; }

    // row-major bf16
    __bf16* dstb = xb + ((size_t)b * S_ + s) * D_ + dbase + cs;
    *(v8bf*)dstb = v0;  *(v8bf*)(dstb + 8) = v1;

    // stage transposed in LDS: t[d_local][i_local]
    #pragma unroll
    for (int j = 0; j < 8; ++j) { t[cs + j][row] = v0[j]; t[cs + 8 + j][row] = v1[j]; }
    __syncthreads();

    // write xt[d][i] rows (contiguous in i)
    __bf16* dstt = xt + ((size_t)bo * D_ + dbase + row) * NSL_ + ibase + cs;
    *(v8bf*)dstt       = *(const v8bf*)&t[row][cs];
    *(v8bf*)(dstt + 8) = *(const v8bf*)&t[row][cs + 8];
}

// ======================= fast attention kernel ===============================
__global__ __launch_bounds__(256)
void dilattn_fast(const __bf16* __restrict__ xb, const __bf16* __restrict__ xt,
                  float* __restrict__ out) {
    __shared__ __bf16 pm[32][PM_LD];
    __shared__ float  redm[8][32];
    __shared__ float  redl[8][32];
    __shared__ float  rowmax[32];
    __shared__ float  rowinv[32];

    const int tid  = threadIdx.x;
    const int wave = tid >> 5;
    const int lane = tid & 31;
    const int l16  = lane & 15;
    const int hi   = lane >> 4;

    const int blk   = blockIdx.x;
    const int prob  = blk >> 4;                  // 0..63
    const int qtile = blk & 15;
    const int b     = prob >> 4;
    const int rem   = prob & 15;
    const int off   = rem >> 2;
    const int gseg  = rem & 3;
    const int qbase = qtile * 32;

    const size_t probOfs = ((size_t)b * S_ + (size_t)(gseg * SEG_) * R_ + off) * D_;
    const __bf16* __restrict__ base = xb  + probOfs;
    float*        __restrict__ obas = out + probOfs;

    // ---------------- Phase 1: S = Q K^T (registers) -----------------------
    v8f acc[2][4];
    #pragma unroll
    for (int qs = 0; qs < 2; ++qs)
        #pragma unroll
        for (int t = 0; t < 4; ++t) acc[qs][t] = (v8f){};
    {
        const __bf16* qrow0 = base + (size_t)(qbase + l16) * ROWSTRIDE;
        const __bf16* qrow1 = qrow0 + (size_t)16 * ROWSTRIDE;
        const __bf16* krow[4];
        #pragma unroll
        for (int t = 0; t < 4; ++t)
            krow[t] = base + (size_t)(wave * 64 + t * 16 + l16) * ROWSTRIDE;

        for (int dt = 0; dt < 32; ++dt) {
            const int d0 = dt * 32 + hi * 8;
            const v16bf af0 = ldfrag(qrow0 + d0);
            const v16bf af1 = ldfrag(qrow1 + d0);
            #pragma unroll
            for (int t = 0; t < 4; ++t) {
                const v16bf bfm = ldfrag(krow[t] + d0);
                acc[0][t] = WMMA_BF16(af0, bfm, acc[0][t]);
                acc[1][t] = WMMA_BF16(af1, bfm, acc[1][t]);
            }
        }
    }

    // ---------------- Phase 2: softmax from registers ----------------------
    #pragma unroll
    for (int qs = 0; qs < 2; ++qs)
        #pragma unroll
        for (int i = 0; i < 8; ++i) {
            float m = acc[qs][0][i];
            #pragma unroll
            for (int t = 1; t < 4; ++t) m = fmaxf(m, acc[qs][t][i]);
            m = swz_max16(m);
            if (l16 == 0) redm[wave][qs * 16 + i + 8 * hi] = m;
        }
    __syncthreads();
    if (tid < 32) {
        float m = redm[0][tid];
        #pragma unroll
        for (int w = 1; w < 8; ++w) m = fmaxf(m, redm[w][tid]);
        rowmax[tid] = m;
    }
    __syncthreads();
    #pragma unroll
    for (int qs = 0; qs < 2; ++qs)
        #pragma unroll
        for (int i = 0; i < 8; ++i) {
            const int q = qs * 16 + i + 8 * hi;
            const float rm = rowmax[q];
            float s = 0.f;
            #pragma unroll
            for (int t = 0; t < 4; ++t) {
                const float p = __expf((acc[qs][t][i] - rm) * SCALE);
                s += p;
                pm[q][wave * 64 + t * 16 + l16] = (__bf16)p;
            }
            s = swz_sum16(s);
            if (l16 == 0) redl[wave][q] = s;
        }
    __syncthreads();
    if (tid < 32) {
        float s = redl[0][tid];
        #pragma unroll
        for (int w = 1; w < 8; ++w) s += redl[w][tid];
        rowinv[tid] = 0.25f / s;                 // fold 1/r
    }
    __syncthreads();

    // -------- Phase 3: O^T = V^T P^T ; wave owns 128 d-rows of V^T ---------
    {
        const __bf16* vt = xt + ((size_t)(b * 4 + off) * D_) * NSL_ + (size_t)gseg * SEG_;
        v8f o[8][2];
        #pragma unroll
        for (int m = 0; m < 8; ++m) { o[m][0] = (v8f){}; o[m][1] = (v8f){}; }

        for (int kt = 0; kt < 16; ++kt) {        // 512 keys / 32
            const int k0 = kt * 32 + hi * 8;
            const v16bf pf0 = ldfrag(&pm[l16][k0]);        // P^T B-fragment, q 0-15
            const v16bf pf1 = ldfrag(&pm[16 + l16][k0]);   // q 16-31
            #pragma unroll
            for (int m = 0; m < 8; ++m) {
                const int dm = wave * 128 + m * 16;
                const v16bf afm = ldfrag(vt + (size_t)(dm + l16) * NSL_ + k0);
                o[m][0] = WMMA_BF16(afm, pf0, o[m][0]);
                o[m][1] = WMMA_BF16(afm, pf1, o[m][1]);
            }
        }

        // O^T tile: lane&15 = q_local (N), VGPR i -> d_local = i + 8*hi (M)
        const float fac0 = rowinv[l16];
        const float fac1 = rowinv[16 + l16];
        float* orow0 = obas + (size_t)(qbase + l16) * ROWSTRIDE;
        float* orow1 = obas + (size_t)(qbase + 16 + l16) * ROWSTRIDE;
        #pragma unroll
        for (int m = 0; m < 8; ++m) {
            const int dm = wave * 128 + m * 16 + 8 * hi;
            float4 a0 = make_float4(o[m][0][0] * fac0, o[m][0][1] * fac0,
                                    o[m][0][2] * fac0, o[m][0][3] * fac0);
            float4 a1 = make_float4(o[m][0][4] * fac0, o[m][0][5] * fac0,
                                    o[m][0][6] * fac0, o[m][0][7] * fac0);
            float4 b0 = make_float4(o[m][1][0] * fac1, o[m][1][1] * fac1,
                                    o[m][1][2] * fac1, o[m][1][3] * fac1);
            float4 b1 = make_float4(o[m][1][4] * fac1, o[m][1][5] * fac1,
                                    o[m][1][6] * fac1, o[m][1][7] * fac1);
            *(float4*)(orow0 + dm)     = a0;
            *(float4*)(orow0 + dm + 4) = a1;
            *(float4*)(orow1 + dm)     = b0;
            *(float4*)(orow1 + dm + 4) = b1;
        }
    }
}

// ======================= fallback (no workspace) =============================
__global__ __launch_bounds__(256)
void dilattn_fallback(const float* __restrict__ x, float* __restrict__ out) {
    __shared__ __bf16 pm[32][PM_LD];
    __shared__ float  redm[8][32];
    __shared__ float  redl[8][32];
    __shared__ float  rowmax[32];
    __shared__ float  rowinv[32];

    const int tid  = threadIdx.x;
    const int wave = tid >> 5;
    const int lane = tid & 31;
    const int l16  = lane & 15;
    const int hi   = lane >> 4;

    const int blk   = blockIdx.x;
    const int prob  = blk >> 4;
    const int qtile = blk & 15;
    const int b     = prob >> 4;
    const int rem   = prob & 15;
    const int off   = rem >> 2;
    const int gseg  = rem & 3;
    const int qbase = qtile * 32;

    const size_t probOfs = ((size_t)b * S_ + (size_t)(gseg * SEG_) * R_ + off) * D_;
    const float* __restrict__ base = x   + probOfs;
    float*       __restrict__ obas = out + probOfs;

    v8f acc[2][4];
    #pragma unroll
    for (int qs = 0; qs < 2; ++qs)
        #pragma unroll
        for (int t = 0; t < 4; ++t) acc[qs][t] = (v8f){};
    {
        const float* qrow0 = base + (size_t)(qbase + l16) * ROWSTRIDE;
        const float* qrow1 = qrow0 + (size_t)16 * ROWSTRIDE;
        const float* krow[4];
        #pragma unroll
        for (int t = 0; t < 4; ++t)
            krow[t] = base + (size_t)(wave * 64 + t * 16 + l16) * ROWSTRIDE;
        for (int dt = 0; dt < 32; ++dt) {
            const int d0 = dt * 32 + hi * 8;
            v16bf af0, af1;
            #pragma unroll
            for (int j = 0; j < 8; ++j) {
                af0[j] = (__bf16)qrow0[d0 + j]; af0[j + 8] = (__bf16)qrow0[d0 + 16 + j];
                af1[j] = (__bf16)qrow1[d0 + j]; af1[j + 8] = (__bf16)qrow1[d0 + 16 + j];
            }
            #pragma unroll
            for (int t = 0; t < 4; ++t) {
                v16bf bfm;
                #pragma unroll
                for (int j = 0; j < 8; ++j) {
                    bfm[j] = (__bf16)krow[t][d0 + j]; bfm[j + 8] = (__bf16)krow[t][d0 + 16 + j];
                }
                acc[0][t] = WMMA_BF16(af0, bfm, acc[0][t]);
                acc[1][t] = WMMA_BF16(af1, bfm, acc[1][t]);
            }
        }
    }
    #pragma unroll
    for (int qs = 0; qs < 2; ++qs)
        #pragma unroll
        for (int i = 0; i < 8; ++i) {
            float m = acc[qs][0][i];
            #pragma unroll
            for (int t = 1; t < 4; ++t) m = fmaxf(m, acc[qs][t][i]);
            m = swz_max16(m);
            if (l16 == 0) redm[wave][qs * 16 + i + 8 * hi] = m;
        }
    __syncthreads();
    if (tid < 32) {
        float m = redm[0][tid];
        #pragma unroll
        for (int w = 1; w < 8; ++w) m = fmaxf(m, redm[w][tid]);
        rowmax[tid] = m;
    }
    __syncthreads();
    #pragma unroll
    for (int qs = 0; qs < 2; ++qs)
        #pragma unroll
        for (int i = 0; i < 8; ++i) {
            const int q = qs * 16 + i + 8 * hi;
            const float rm = rowmax[q];
            float s = 0.f;
            #pragma unroll
            for (int t = 0; t < 4; ++t) {
                const float p = __expf((acc[qs][t][i] - rm) * SCALE);
                s += p;
                pm[q][wave * 64 + t * 16 + l16] = (__bf16)p;
            }
            s = swz_sum16(s);
            if (l16 == 0) redl[wave][q] = s;
        }
    __syncthreads();
    if (tid < 32) {
        float s = redl[0][tid];
        #pragma unroll
        for (int w = 1; w < 8; ++w) s += redl[w][tid];
        rowinv[tid] = 0.25f / s;
    }
    __syncthreads();
    {
        const int dbase = wave * 128;
        v8f o[2][8];
        #pragma unroll
        for (int qs = 0; qs < 2; ++qs)
            #pragma unroll
            for (int nt = 0; nt < 8; ++nt) o[qs][nt] = (v8f){};
        for (int kt = 0; kt < 16; ++kt) {
            const int k0 = kt * 32 + hi * 8;
            const v16bf pf0 = ldfrag(&pm[l16][k0]);
            const v16bf pf1 = ldfrag(&pm[16 + l16][k0]);
            #pragma unroll
            for (int nt = 0; nt < 8; ++nt) {
                const int dcol = dbase + nt * 16 + l16;
                const float* vcol = base + dcol + (size_t)k0 * ROWSTRIDE;
                v16bf vf;
                #pragma unroll
                for (int j = 0; j < 8; ++j) {
                    vf[j]     = (__bf16)vcol[(size_t)j * ROWSTRIDE];
                    vf[j + 8] = (__bf16)vcol[(size_t)(j + 16) * ROWSTRIDE];
                }
                o[0][nt] = WMMA_BF16(pf0, vf, o[0][nt]);
                o[1][nt] = WMMA_BF16(pf1, vf, o[1][nt]);
            }
        }
        float rw[2][8];
        #pragma unroll
        for (int qs = 0; qs < 2; ++qs)
            #pragma unroll
            for (int i = 0; i < 8; ++i)
                rw[qs][i] = rowinv[qs * 16 + i + 8 * hi];
        #pragma unroll
        for (int nt = 0; nt < 8; ++nt) {
            const int dcol = dbase + nt * 16 + l16;
            #pragma unroll
            for (int qs = 0; qs < 2; ++qs) {
                float* orow = obas + dcol + (size_t)(qbase + qs * 16 + 8 * hi) * ROWSTRIDE;
                #pragma unroll
                for (int i = 0; i < 8; ++i)
                    orow[(size_t)i * ROWSTRIDE] = o[qs][nt][i] * rw[qs][i];
            }
        }
    }
}

extern "C" void kernel_launch(void* const* d_in, const int* in_sizes, int n_in,
                              void* d_out, int out_size, void* d_ws, size_t ws_size,
                              hipStream_t stream) {
    (void)in_sizes; (void)n_in; (void)out_size;
    const float* x = (const float*)d_in[0];
    float* out = (float*)d_out;

    const size_t nElems = (size_t)B_ * S_ * D_;          // 33.5M
    const size_t need   = 2 * nElems * sizeof(__bf16);   // 128 MiB (xb + xt)

    if (d_ws != nullptr && ws_size >= need) {
        __bf16* xb = (__bf16*)d_ws;
        __bf16* xt = xb + nElems;
        prepass_kernel<<<dim3(8192), dim3(256), 0, stream>>>(x, xb, xt);
        dilattn_fast<<<dim3(64 * 16), dim3(256), 0, stream>>>(xb, xt, out);
    } else {
        dilattn_fallback<<<dim3(64 * 16), dim3(256), 0, stream>>>(x, out);
    }
}
